// VectorQuantizer_6141803233313
// MI455X (gfx1250) — matrix operand order repaired and checked by reading, not compiled, not run on hardware
//
#include <hip/hip_runtime.h>

typedef __attribute__((ext_vector_type(16))) __bf16 v16bf;
typedef __attribute__((ext_vector_type(8)))  __bf16 v8bf;
typedef __attribute__((ext_vector_type(8)))  float  v8f;

#define K_CODES 1024
#define DIM 64
#define ROW_STRIDE 72      // bf16 elements per LDS codebook row (64 + 8 pad -> conflict-free)
#define TILES (K_CODES/16)

#if defined(__HIP_DEVICE_COMPILE__) && __has_builtin(__builtin_amdgcn_global_load_async_to_lds_b128)
#define VQ_ASYNC_LDS 1
typedef __attribute__((ext_vector_type(4))) int i32x4;
typedef __attribute__((address_space(1))) i32x4 g_i32x4;   // global int4
typedef __attribute__((address_space(3))) i32x4 l_i32x4;   // LDS int4
#endif

// ---- prep: split fp32 codebook into bf16 hi/lo + 0.5*||e||^2 -----------
__global__ void vq_prep(const float* __restrict__ cb,
                        __bf16* __restrict__ hi, __bf16* __restrict__ lo,
                        float* __restrict__ halfn)
{
  __shared__ float red[DIM];
  const int c = blockIdx.x, t = threadIdx.x;
  float x = cb[c*DIM + t];
  __bf16 h = (__bf16)x;
  __bf16 l = (__bf16)(x - (float)h);
  hi[c*DIM + t] = h;
  lo[c*DIM + t] = l;
  red[t] = x * x;
  __syncthreads();
  for (int s = DIM/2; s > 0; s >>= 1) {
    if (t < s) red[t] += red[t+s];
    __syncthreads();
  }
  if (t == 0) halfn[c] = 0.5f * red[0];
}

__global__ void vq_zero(unsigned* __restrict__ counts)
{
  int i = blockIdx.x * blockDim.x + threadIdx.x;
  if (i < K_CODES) counts[i] = 0u;
}

// ---- device helpers ----------------------------------------------------
__device__ __forceinline__ v8f wmma3(v8f acc, v16bf ah, v16bf al, v16bf bh, v16bf bl)
{
  // bf16x3 error-compensated product: hi*hi + hi*lo + lo*hi, f32 accumulate
  acc = __builtin_amdgcn_wmma_f32_16x16x32_bf16(false, ah, false, bh, (short)0, acc, false, false);
  acc = __builtin_amdgcn_wmma_f32_16x16x32_bf16(false, ah, false, bl, (short)0, acc, false, false);
  acc = __builtin_amdgcn_wmma_f32_16x16x32_bf16(false, al, false, bh, (short)0, acc, false, false);
  return acc;
}

// A fragments: 16x32 bf16 per K-step; lane<16 holds K={o..o+7,16+o..16+o+7}, o=half*8
__device__ __forceinline__ void load_A(const float* __restrict__ ztile,
                                       int lrow, int half, v16bf aH[2], v16bf aL[2])
{
#pragma unroll
  for (int ks = 0; ks < 2; ++ks) {
    const float4* pa = (const float4*)(ztile + lrow*DIM + ks*32 + half*8);
    const float4* pb = (const float4*)(ztile + lrow*DIM + ks*32 + 16 + half*8);
    float4 a0 = pa[0], a1 = pa[1], b0 = pb[0], b1 = pb[1];
    float ga[8] = {a0.x,a0.y,a0.z,a0.w,a1.x,a1.y,a1.z,a1.w};
    float gb[8] = {b0.x,b0.y,b0.z,b0.w,b1.x,b1.y,b1.z,b1.w};
#pragma unroll
    for (int j = 0; j < 8; ++j) {
      __bf16 h0 = (__bf16)ga[j], h1 = (__bf16)gb[j];
      aH[ks][j]   = h0;  aH[ks][8+j] = h1;
      aL[ks][j]   = (__bf16)(ga[j] - (float)h0);
      aL[ks][8+j] = (__bf16)(gb[j] - (float)h1);
    }
  }
}

// B fragments for one 16-code tile: lane holds column lrow, K={o..o+7,16+o..16+o+7}, o=half*8 (+ks*32)
__device__ __forceinline__ void load_B(const __bf16* __restrict__ sHI,
                                       const __bf16* __restrict__ sLO,
                                       int t, int lrow, int half,
                                       v16bf Bh[2], v16bf Bl[2])
{
  const __bf16* rh = sHI + (t*16 + lrow)*ROW_STRIDE + half*8;
  const __bf16* rl = sLO + (t*16 + lrow)*ROW_STRIDE + half*8;
#pragma unroll
  for (int ks = 0; ks < 2; ++ks) {
    v8bf h0 = *(const v8bf*)(rh + ks*32),  h1 = *(const v8bf*)(rh + ks*32 + 16);
    v8bf l0 = *(const v8bf*)(rl + ks*32),  l1 = *(const v8bf*)(rl + ks*32 + 16);
    Bh[ks] = __builtin_shufflevector(h0, h1, 0,1,2,3,4,5,6,7,8,9,10,11,12,13,14,15);
    Bl[ks] = __builtin_shufflevector(l0, l1, 0,1,2,3,4,5,6,7,8,9,10,11,12,13,14,15);
  }
}

// Score one 16-code tile against two 16-row A tiles; update per-slot argmax.
__device__ __forceinline__ void tile_scores(const v16bf aH0[2], const v16bf aL0[2],
                                            const v16bf aH1[2], const v16bf aL1[2],
                                            const v16bf Bh[2],  const v16bf Bl[2],
                                            const float* __restrict__ sHN,
                                            int t, int lrow,
                                            float bv0[8], int bi0[8],
                                            float bv1[8], int bi1[8])
{
  v8f acc0 = {}, acc1 = {};
#pragma unroll
  for (int ks = 0; ks < 2; ++ks) {
    acc0 = wmma3(acc0, aH0[ks], aL0[ks], Bh[ks], Bl[ks]);
    acc1 = wmma3(acc1, aH1[ks], aL1[ks], Bh[ks], Bl[ks]);
  }
  const int   cidx = t*16 + lrow;
  const float hn   = sHN[cidx];
#pragma unroll
  for (int v = 0; v < 8; ++v) {
    float s0 = acc0[v] - hn;
    if (s0 > bv0[v]) { bv0[v] = s0; bi0[v] = cidx; }
    float s1 = acc1[v] - hn;
    if (s1 > bv1[v]) { bv1[v] = s1; bi1[v] = cidx; }
  }
}

// ---- main: GEMM-argmax via bf16x3 WMMA, gather, SSE, counts ------------
__global__ __launch_bounds__(256)
void vq_main(const float* __restrict__ z, const float* __restrict__ cb,
             const __bf16* __restrict__ cb_hi, const __bf16* __restrict__ cb_lo,
             const float* __restrict__ halfn,
             float* __restrict__ out_zq, float* __restrict__ partials,
             unsigned* __restrict__ counts, int n_rowtiles)
{
  extern __shared__ char smem[];
  __bf16* sHI  = (__bf16*)smem;                         // 1024 x 72 bf16
  __bf16* sLO  = sHI + K_CODES*ROW_STRIDE;              // 1024 x 72 bf16
  float*  sHN  = (float*)(sLO + K_CODES*ROW_STRIDE);    // 1024 f32
  int*    sIDX = (int*)(sHN + K_CODES);                 // 8 waves x 32 rows

  const int tid  = threadIdx.x;
  const int wave = tid >> 5;
  const int lane = tid & 31;
  const int lrow = lane & 15;     // row (A) or column (B, C) within 16-tile
  const int half = lane >> 4;     // half-wave select per ISA layouts

  // ---- stage hi/lo bf16 codebook into LDS (320KB/WGP) ----
#ifdef VQ_ASYNC_LDS
  // gfx1250 async global->LDS copies, tracked by ASYNCcnt
  for (int r = tid; r < K_CODES; r += 256) {
    const char* srcH = (const char*)(cb_hi + r*DIM);
    const char* srcL = (const char*)(cb_lo + r*DIM);
    char* dH = (char*)(sHI + r*ROW_STRIDE);
    char* dL = (char*)(sLO + r*ROW_STRIDE);
#pragma unroll
    for (int j = 0; j < 8; ++j) {
      __builtin_amdgcn_global_load_async_to_lds_b128((g_i32x4*)(srcH + j*16),
                                                     (l_i32x4*)(dH + j*16), 0, 0);
      __builtin_amdgcn_global_load_async_to_lds_b128((g_i32x4*)(srcL + j*16),
                                                     (l_i32x4*)(dL + j*16), 0, 0);
    }
  }
#if __has_builtin(__builtin_amdgcn_s_wait_asynccnt)
  __builtin_amdgcn_s_wait_asynccnt(0);
#else
  asm volatile("s_wait_asynccnt 0x0" ::: "memory");
#endif
#else
  for (int r = tid; r < K_CODES; r += 256) {
    const v8bf* sh = (const v8bf*)(cb_hi + r*DIM);
    const v8bf* sl = (const v8bf*)(cb_lo + r*DIM);
    v8bf* dh = (v8bf*)(sHI + r*ROW_STRIDE);
    v8bf* dl = (v8bf*)(sLO + r*ROW_STRIDE);
#pragma unroll
    for (int j = 0; j < 8; ++j) { dh[j] = sh[j]; dl[j] = sl[j]; }
  }
#endif
  for (int k = tid; k < K_CODES; k += 256) sHN[k] = halfn[k];
  __syncthreads();

  const int gwave   = blockIdx.x*8 + wave;
  const int nwaves  = gridDim.x*8;
  const int n_pairs = n_rowtiles >> 1;   // each wave does 2 row-tiles (32 rows)
  float wave_sse = 0.f;

  for (int rp = gwave; rp < n_pairs; rp += nwaves) {
    const float* zt0 = z + (size_t)rp * 32 * DIM;
    const float* zt1 = zt0 + 16*DIM;

    // prefetch next row-pair of z (global_prefetch_b8)
    int rpn = rp + nwaves;
    if (rpn < n_pairs)
      __builtin_prefetch((const char*)(z + (size_t)rpn*32*DIM) + lane*256, 0, 1);

    v16bf aH0[2], aL0[2], aH1[2], aL1[2];
    load_A(zt0, lrow, half, aH0, aL0);
    load_A(zt1, lrow, half, aH1, aL1);

    float bv0[8], bv1[8]; int bi0[8], bi1[8];
#pragma unroll
    for (int v = 0; v < 8; ++v) {
      bv0[v] = -3.4e38f; bi0[v] = 0;
      bv1[v] = -3.4e38f; bi1[v] = 0;
    }

    // software-pipelined B double buffer: load tile t+1 while computing t
    v16bf BhA[2], BlA[2], BhB[2], BlB[2];
    load_B(sHI, sLO, 0, lrow, half, BhA, BlA);
    for (int t = 0; t < TILES; t += 2) {
      load_B(sHI, sLO, t+1, lrow, half, BhB, BlB);
      tile_scores(aH0, aL0, aH1, aL1, BhA, BlA, sHN, t,   lrow, bv0, bi0, bv1, bi1);
      if (t + 2 < TILES)
        load_B(sHI, sLO, t+2, lrow, half, BhA, BlA);
      tile_scores(aH0, aL0, aH1, aL1, BhB, BlB, sHN, t+1, lrow, bv0, bi0, bv1, bi1);
    }

    // argmax across the 16 lanes of each half (ties -> lowest code index)
#pragma unroll
    for (int v = 0; v < 8; ++v) {
      float b0 = bv0[v]; int i0 = bi0[v];
      float b1 = bv1[v]; int i1 = bi1[v];
#pragma unroll
      for (int m = 8; m > 0; m >>= 1) {
        float o0 = __shfl_xor(b0, m, 16);  int oi0 = __shfl_xor(i0, m, 16);
        float o1 = __shfl_xor(b1, m, 16);  int oi1 = __shfl_xor(i1, m, 16);
        if (o0 > b0 || (o0 == b0 && oi0 < i0)) { b0 = o0; i0 = oi0; }
        if (o1 > b1 || (o1 == b1 && oi1 < i1)) { b1 = o1; i1 = oi1; }
      }
      if (lrow == 0) {
        sIDX[wave*32 +      half*8 + v] = i0;
        sIDX[wave*32 + 16 + half*8 + v] = i1;
      }
    }

    // gather z_q, write out, accumulate SSE; lane covers (row=lrow, 32 cols)
    float sse = 0.f;
#pragma unroll
    for (int g = 0; g < 2; ++g) {
      const float* ztg = g ? zt1 : zt0;
      const int code = sIDX[wave*32 + g*16 + lrow];
      if (half == 0) atomicAdd(&counts[code], 1u);   // exact integer atomic
      const float4* q4 = (const float4*)(cb + code*DIM) + half*8;
      const float4* z4 = (const float4*)(ztg + lrow*DIM) + half*8;
      float4* o4 = (float4*)(out_zq + ((size_t)rp*32 + g*16 + lrow)*DIM) + half*8;
#pragma unroll
      for (int j = 0; j < 8; ++j) {
        float4 q = q4[j], e = z4[j];
        o4[j] = q;
        float dx=q.x-e.x, dy=q.y-e.y, dz=q.z-e.z, dw=q.w-e.w;
        sse += dx*dx + dy*dy + dz*dz + dw*dw;
      }
    }
#pragma unroll
    for (int m = 16; m > 0; m >>= 1) sse += __shfl_xor(sse, m, 32);
    wave_sse += sse;
  }
  if (lane == 0) partials[gwave] = wave_sse;   // deterministic: no float atomics
}

// ---- finisher: loss + perplexity ---------------------------------------
__global__ void vq_final(const unsigned* __restrict__ counts,
                         const float* __restrict__ partials, int n_partials,
                         float* __restrict__ out2, float invN, float lossScale)
{
  __shared__ float redp[1024];
  __shared__ float reds[1024];
  const int t = threadIdx.x;
  float p = (float)counts[t] * invN;
  redp[t] = p * logf(p + 1e-10f);
  float s = 0.f;
  for (int i = t; i < n_partials; i += 1024) s += partials[i];
  reds[t] = s;
  __syncthreads();
  for (int k = 512; k > 0; k >>= 1) {
    if (t < k) { redp[t] += redp[t+k]; reds[t] += reds[t+k]; }
    __syncthreads();
  }
  if (t == 0) {
    out2[0] = reds[0] * lossScale;   // loss = 1.25 * SSE / (N*D)
    out2[1] = expf(-redp[0]);        // perplexity
  }
}

extern "C" void kernel_launch(void* const* d_in, const int* in_sizes, int n_in,
                              void* d_out, int out_size, void* d_ws, size_t ws_size,
                              hipStream_t stream)
{
  const float* z  = (const float*)d_in[0];   // z_e  [64,4096,64] f32
  const float* cb = (const float*)d_in[1];   // codebook [1024,64] f32
  float* out = (float*)d_out;                // [z_q | loss | perplexity]

  const int n_elem     = in_sizes[0];        // 16,777,216
  const int N          = n_elem / DIM;       // 262,144 vectors
  const int n_rowtiles = N / 16;             // 16,384

  char* ws = (char*)d_ws;
  unsigned* counts   = (unsigned*)(ws);                  // 4 KB
  float*    halfn    = (float*)(ws + 4096);              // 4 KB
  float*    partials = (float*)(ws + 8192);              // 16 KB
  __bf16*   cb_hi    = (__bf16*)(ws + 24576);            // 128 KB
  __bf16*   cb_lo    = (__bf16*)(ws + 24576 + 131072);   // 128 KB

  const int grid = 512;                 // grid-stride; 8 waves/block
  const int n_partials = grid * 8;

  vq_prep<<<K_CODES, DIM, 0, stream>>>(cb, cb_hi, cb_lo, halfn);
  vq_zero<<<K_CODES/256, 256, 0, stream>>>(counts);

  size_t smem = (size_t)K_CODES*ROW_STRIDE*2*sizeof(__bf16)   // hi+lo codebook
              + (size_t)K_CODES*sizeof(float)                 // half norms
              + 8*32*sizeof(int);                             // per-wave idx
  vq_main<<<grid, 256, smem, stream>>>(z, cb, cb_hi, cb_lo, halfn,
                                       out, partials, counts, n_rowtiles);

  vq_final<<<1, 1024, 0, stream>>>(counts, partials, n_partials,
                                   out + (size_t)n_elem,
                                   1.0f / (float)N,
                                   1.25f / (float)n_elem);
}
